// Repro_15075335209179
// MI455X (gfx1250) — compile-verified, matrix operand order checked by reading, not run on hardware
//
#include <hip/hip_runtime.h>
#include <hip/hip_bf16.h>
#include <math.h>

// ---- problem constants (match reference) ----
#define BATCH 500
#define SEQL 128
#define FIN  16
#define DM   128
#define NH   8
#define HD   16
#define DFF  512
#define DROP_P  0.1f
#define DROP_SC 1.1111111111111112f

typedef __attribute__((ext_vector_type(16))) _Float16 v16h;
typedef __attribute__((ext_vector_type(8)))  _Float16 v8h;
typedef __attribute__((ext_vector_type(4)))  _Float16 v4h;
typedef __attribute__((ext_vector_type(8)))  float    v8f;
typedef __attribute__((ext_vector_type(4)))  float    v4f;

// ---------------- WMMA fragment helpers (CDNA5 wave32 layouts) ----------------
// For a 16x32 f16 A/B fragment, lane = (m|n) + 16*hb holds K pairs at
// k = hb*8 + g*16 + q*2 (+1): element j (0..15) of the v16h maps to
// k = hb*8 + (j>>3)*16 + (j&7): two CONTIGUOUS 8-half (16B) groups.
__device__ __forceinline__ v16h ldA16(const _Float16* s, int ld, int m0, int k0, int lane) {
  const _Float16* row = s + (size_t)(m0 + (lane & 15)) * ld + k0 + ((lane >> 4) << 3);
  v8h lo = *(const v8h*)(row);
  v8h hi = *(const v8h*)(row + 16);
  return __builtin_shufflevector(lo, hi, 0,1,2,3,4,5,6,7,8,9,10,11,12,13,14,15);
}

// Head fragment: logical K=16 (head dim), padded to K=32 with zeros.
__device__ __forceinline__ v16h ldHead(const _Float16* s, int ld, int m0, int lane) {
  const _Float16* row = s + (size_t)(m0 + (lane & 15)) * ld + ((lane >> 4) << 3);
  v8h lo = *(const v8h*)(row);
  v8h z = {};
  return __builtin_shufflevector(lo, z, 0,1,2,3,4,5,6,7,8,9,10,11,12,13,14,15);
}

__device__ __forceinline__ v8f wmma32(v16h a, v16h b, v8f c) {
  return __builtin_amdgcn_wmma_f32_16x16x32_f16(false, a, false, b, (short)0, c, false, false);
}

// Cooperative f32 -> f16 weight staging (contiguous src), v4f in / v4h out.
__device__ __forceinline__ void stage_w16(const float* __restrict__ src, _Float16* dst,
                                          int nElem, int tid) {
  const v4f* s4 = (const v4f*)src;
  int nQuad = nElem >> 2;
  for (int i = tid; i < nQuad; i += 256) {
    v4f w = s4[i];
    v4h p;
#pragma unroll
    for (int j = 0; j < 4; ++j) p[j] = (_Float16)w[j];
    *(v4h*)(dst + (size_t)i * 4) = p;
  }
}

// ---------------- fused transformer-block kernel: 1 workgroup per batch elem ----------------
__global__ __launch_bounds__(256) void fused_block(
    const float* __restrict__ x,       const float* __restrict__ W_embed,
    const float* __restrict__ b_embed, const float* __restrict__ g0,
    const float* __restrict__ b0,      const float* __restrict__ W_qkv,
    const float* __restrict__ b_qkv,   const float* __restrict__ attn_bias,
    const float* __restrict__ key_bias,const float* __restrict__ W_out,
    const float* __restrict__ b_out,   const float* __restrict__ g1,
    const float* __restrict__ b1,      const float* __restrict__ g2,
    const float* __restrict__ b2,      const float* __restrict__ W_ff1,
    const float* __restrict__ b_ff1,   const float* __restrict__ W_ff2,
    const float* __restrict__ b_ff2,   const float* __restrict__ m_attn,
    const float* __restrict__ m1,      const float* __restrict__ m_ff,
    const float* __restrict__ m2,      float* __restrict__ out) {
  const int b    = blockIdx.x;
  const int tid  = threadIdx.x;
  const int lane = tid & 31;
  const int wave = tid >> 5;

  __shared__ float sInv[SEQL];     // softmax 1/sum per row
  __shared__ float sMean[SEQL];    // LN1 stats
  __shared__ float sRs[SEQL];

  extern __shared__ char smem[];
  float*    sH0  = (float*)(smem);                 // [128][128] f32: h0, later y (residual)
  _Float16* sAct = (_Float16*)(smem + 65536);      // [128][128] f16: GEMM A operand
  char*     big  = smem + 98304;                   // 196608-byte multiplexed region
  // attention-phase layout
  _Float16* sQ   = (_Float16*)big;                 // [128][128] f16 (q * 0.25), col = h*16+j
  _Float16* sK   = (_Float16*)(big + 32768);
  _Float16* sVt  = (_Float16*)(big + 65536);       // V TRANSPOSED: [d=128][row=128]
  float*    sS   = (float*)(big + 98304);          // [128][128] f32 scores / out-proj o
  _Float16* sP   = (_Float16*)(big + 163840);      // [128][128] f16 probs
  // staged-weight aliases
  _Float16* sWq  = (_Float16*)(big + 98304);       // W_qkv f16 [384][128] (pre-attention)
  _Float16* sWo  = (_Float16*)(big + 163840);      // W_out f16 [128][128] (post-attention)
  _Float16* sWst = (_Float16*)(big + 131072);      // 64KB weight staging for FFN halves
  // FFN-phase aliases
  _Float16* sF   = (_Float16*)big;                 // [128][512] f16 FFN hidden
  float*    sX   = (float*)big;                    // [128][16]  f32 x staging

  // ---- Stage A: copy x (NT v4f) + stage W_qkv f16; h0 = x @ W_embed^T + b_embed ----
  {
    const v4f* x4 = (const v4f*)(x + (size_t)b * SEQL * FIN);
    v4f* sX4 = (v4f*)sX;
    for (int i = tid; i < (SEQL * FIN) / 4; i += 256)
      sX4[i] = __builtin_nontemporal_load(x4 + i);
  }
  stage_w16(W_qkv, sWq, 3 * DM * DM, tid);
  __syncthreads();
  for (int i = tid; i < SEQL * DM; i += 256) {
    int r = i >> 7, d = i & 127;
    const float* xr = sX + r * FIN;
    const float* wr = W_embed + d * FIN;
    float acc = b_embed[d];
#pragma unroll
    for (int f = 0; f < FIN; ++f) acc += xr[f] * wr[f];
    sH0[i] = acc;
  }
  __syncthreads();

  // ---- Stage B: LN0 -> sAct (f16), vectorized row scan ----
  if (tid < SEQL) {
    const v4f* hr4 = (const v4f*)(sH0 + tid * DM);
    v4f s4 = {};
    for (int q = 0; q < 32; ++q) s4 += hr4[q];
    float mean = (s4[0] + s4[1] + s4[2] + s4[3]) * (1.f / DM);
    v4f v4 = {};
    for (int q = 0; q < 32; ++q) { v4f t = hr4[q] - mean; v4 += t * t; }
    float var = (v4[0] + v4[1] + v4[2] + v4[3]) * (1.f / DM);
    float rs = rsqrtf(var + 1e-5f);
    const v4f* g4 = (const v4f*)g0;
    const v4f* bb4 = (const v4f*)b0;
    v4h* ar4 = (v4h*)(sAct + tid * DM);
    for (int q = 0; q < 32; ++q) {
      v4f t = (hr4[q] - mean) * rs * g4[q] + bb4[q];
      v4h o;
#pragma unroll
      for (int j = 0; j < 4; ++j) o[j] = (_Float16)t[j];
      ar4[q] = o;
    }
  }
  __syncthreads();

  // ---- Stage C: qkv = LN0 @ W_qkv^T + b_qkv -> sQ/sK/sVt (head layout, V transposed) ----
  for (int t = wave; t < 8 * 24; t += 8) {
    int m0 = (t / 24) << 4, n0 = (t % 24) << 4;
    v8f c = {};
#pragma unroll
    for (int k0 = 0; k0 < DM; k0 += 32)
      c = wmma32(ldA16(sAct, DM, m0, k0, lane), ldA16(sWq, DM, n0, k0, lane), c);
    int n = n0 + (lane & 15);
    int part = n >> 7, d = n & 127;
    float bias = b_qkv[n];
    int rb = m0 + ((lane >> 4) << 3);
    if (part == 2) {                       // V: transposed, contiguous 8-half store
      v8h pk;
#pragma unroll
      for (int v = 0; v < 8; ++v) pk[v] = (_Float16)(c[v] + bias);
      *(v8h*)(sVt + (size_t)d * SEQL + rb) = pk;
    } else {
      _Float16* dst = (part == 0 ? sQ : sK) + d;
      float sc = (part == 0) ? 0.25f : 1.0f;   // fold 1/sqrt(hd) into q
#pragma unroll
      for (int v = 0; v < 8; ++v) dst[(rb + v) * DM] = (_Float16)((c[v] + bias) * sc);
    }
  }
  __syncthreads();

  // ---- Stage D: attention, one head at a time ----
  const float* kbb = key_bias + (size_t)b * SEQL;
  for (int h = 0; h < NH; ++h) {
    const _Float16* Qh  = sQ + h * HD;
    const _Float16* Kh  = sK + h * HD;
    const _Float16* Vth = sVt + (size_t)h * HD * SEQL;

    // Prefetch this head's m_attn slice into registers (NT): the scores GEMM +
    // softmax below hide the full HBM latency of the dominant 262MB stream.
    const v4f* mh4 = (const v4f*)(m_attn + ((size_t)b * NH + h) * SEQL * SEQL);
    v4f mreg[16];
#pragma unroll
    for (int j = 0; j < 16; ++j)
      mreg[j] = __builtin_nontemporal_load(mh4 + tid + 256 * j);

    // S = Qh @ Kh^T + attn_bias + key_bias   (K=16 padded to 32, 1 WMMA/tile)
    for (int t = wave; t < 64; t += 8) {
      int m0 = (t >> 3) << 4, n0 = (t & 7) << 4;
      v8f c = {};
      c = wmma32(ldHead(Qh, DM, m0, lane), ldHead(Kh, DM, n0, lane), c);
      int n = n0 + (lane & 15);
      int rb = m0 + ((lane >> 4) << 3);
      float kb = kbb[n];
#pragma unroll
      for (int v = 0; v < 8; ++v)
        sS[(rb + v) * SEQL + n] = c[v] + attn_bias[(rb + v) * SEQL + n] + kb;
    }
    __syncthreads();
    // softmax stats per row (exp stored back to sS), vectorized
    if (tid < SEQL) {
      v4f* sr4 = (v4f*)(sS + tid * SEQL);
      v4f mx4 = { -3.4e38f, -3.4e38f, -3.4e38f, -3.4e38f };
      for (int q = 0; q < 32; ++q) {
        v4f s = sr4[q];
#pragma unroll
        for (int j = 0; j < 4; ++j) mx4[j] = fmaxf(mx4[j], s[j]);
      }
      float mx = fmaxf(fmaxf(mx4[0], mx4[1]), fmaxf(mx4[2], mx4[3]));
      v4f sum4 = {};
      for (int q = 0; q < 32; ++q) {
        v4f s = sr4[q], e;
#pragma unroll
        for (int j = 0; j < 4; ++j) e[j] = __expf(s[j] - mx);
        sr4[q] = e;
        sum4 += e;
      }
      sInv[tid] = 1.f / (sum4[0] + sum4[1] + sum4[2] + sum4[3]);
    }
    __syncthreads();
    // apply dropout using the prefetched mask registers -> sP f16 (coalesced)
#pragma unroll
    for (int j = 0; j < 16; ++j) {
      int i = tid + 256 * j;                  // quad index
      float inv = sInv[i >> 5];
      v4f s = *(const v4f*)(sS + 4 * i);
      v4f mk = mreg[j];
      v4h pk;
#pragma unroll
      for (int e = 0; e < 4; ++e)
        pk[e] = (_Float16)((mk[e] > DROP_P) ? s[e] * inv * DROP_SC : 0.f);
      *(v4h*)(sP + 4 * i) = pk;
    }
    __syncthreads();
    // a_h = P @ Vh -> sAct columns [h*16, h*16+16)  (B-fragments from transposed V)
    {
      int m0 = wave << 4;
      v8f c = {};
#pragma unroll
      for (int k0 = 0; k0 < SEQL; k0 += 32)
        c = wmma32(ldA16(sP, SEQL, m0, k0, lane), ldA16(Vth, SEQL, 0, k0, lane), c);
      int j = lane & 15;
      int rb = m0 + ((lane >> 4) << 3);
#pragma unroll
      for (int v = 0; v < 8; ++v) sAct[(rb + v) * DM + h * HD + j] = (_Float16)c[v];
    }
    __syncthreads();
  }

  // ---- Stage E: stage W_out f16; prefetch m1; o = a @ W_out^T + b_out -> sS ----
  stage_w16(W_out, sWo, DM * DM, tid);
  v4f m1reg[16];
  {
    const v4f* m14 = (const v4f*)(m1 + (size_t)b * SEQL * DM);
#pragma unroll
    for (int j = 0; j < 16; ++j)
      m1reg[j] = __builtin_nontemporal_load(m14 + tid + 256 * j);
  }
  __syncthreads();
  for (int t = wave; t < 64; t += 8) {
    int m0 = (t >> 3) << 4, n0 = (t & 7) << 4;
    v8f c = {};
#pragma unroll
    for (int k0 = 0; k0 < DM; k0 += 32)
      c = wmma32(ldA16(sAct, DM, m0, k0, lane), ldA16(sWo, DM, n0, k0, lane), c);
    int n = n0 + (lane & 15);
    float bias = b_out[n];
    int rb = m0 + ((lane >> 4) << 3);
#pragma unroll
    for (int v = 0; v < 8; ++v) sS[(rb + v) * DM + n] = c[v] + bias;
  }
  __syncthreads();
  // LN1 stats (per row), vectorized
  if (tid < SEQL) {
    const v4f* or4 = (const v4f*)(sS + tid * DM);
    v4f s4 = {};
    for (int q = 0; q < 32; ++q) s4 += or4[q];
    float mean = (s4[0] + s4[1] + s4[2] + s4[3]) * (1.f / DM);
    v4f v4 = {};
    for (int q = 0; q < 32; ++q) { v4f t = or4[q] - mean; v4 += t * t; }
    sMean[tid] = mean;
    sRs[tid] = rsqrtf((v4[0] + v4[1] + v4[2] + v4[3]) * (1.f / DM) + 1e-5f);
  }
  __syncthreads();
  // y = drop(LN1(o)) + h0, in place into sH0, using prefetched m1 registers
  {
    const v4f* g14 = (const v4f*)g1;
    const v4f* b14 = (const v4f*)b1;
#pragma unroll
    for (int j = 0; j < 16; ++j) {
      int i = tid + 256 * j;                  // quad index
      int r = i >> 5, qd = i & 31;
      v4f o = *(const v4f*)(sS + 4 * i);
      v4f oln = (o - sMean[r]) * sRs[r] * g14[qd] + b14[qd];
      v4f mk = m1reg[j];
      v4f y = *(const v4f*)(sH0 + 4 * i);
#pragma unroll
      for (int e = 0; e < 4; ++e) y[e] += (mk[e] > DROP_P) ? oln[e] * DROP_SC : 0.f;
      *(v4f*)(sH0 + 4 * i) = y;
    }
  }
  __syncthreads();
  // LN2 per row -> sAct (f16), vectorized
  if (tid < SEQL) {
    const v4f* yr4 = (const v4f*)(sH0 + tid * DM);
    v4f s4 = {};
    for (int q = 0; q < 32; ++q) s4 += yr4[q];
    float mean = (s4[0] + s4[1] + s4[2] + s4[3]) * (1.f / DM);
    v4f v4 = {};
    for (int q = 0; q < 32; ++q) { v4f t = yr4[q] - mean; v4 += t * t; }
    float rs = rsqrtf((v4[0] + v4[1] + v4[2] + v4[3]) * (1.f / DM) + 1e-5f);
    const v4f* g4 = (const v4f*)g2;
    const v4f* bb4 = (const v4f*)b2;
    v4h* ar4 = (v4h*)(sAct + tid * DM);
    for (int q = 0; q < 32; ++q) {
      v4f t = (yr4[q] - mean) * rs * g4[q] + bb4[q];
      v4h o;
#pragma unroll
      for (int j = 0; j < 4; ++j) o[j] = (_Float16)t[j];
      ar4[q] = o;
    }
  }
  __syncthreads();

  // ---- Stage F: f = drop(GELU(y2 @ W_ff1^T + b_ff1)) -> sF [128][512] f16 ----
  for (int nh = 0; nh < 2; ++nh) {
    stage_w16(W_ff1 + (size_t)nh * 256 * DM, sWst, 256 * DM, tid);
    __syncthreads();
    const float* mf = m_ff + (size_t)b * SEQL * DFF;
    for (int t = wave; t < 128; t += 8) {     // 8 row tiles x 16 local col tiles
      int m0 = (t >> 4) << 4, n0 = (t & 15) << 4;
      v8f c = {};
#pragma unroll
      for (int k0 = 0; k0 < DM; k0 += 32)
        c = wmma32(ldA16(sAct, DM, m0, k0, lane), ldA16(sWst, DM, n0, k0, lane), c);
      int n = nh * 256 + n0 + (lane & 15);
      float bias = b_ff1[n];
      int rb = m0 + ((lane >> 4) << 3);
#pragma unroll
      for (int v = 0; v < 8; ++v) {
        int r = rb + v;
        float f = c[v] + bias;
        f = 0.5f * f * (1.0f + erff(f * 0.70710678118654752f));   // exact GELU
        float mk = __builtin_nontemporal_load(mf + (size_t)r * DFF + n);
        sF[(size_t)r * DFF + n] = (_Float16)((mk > DROP_P) ? f * DROP_SC : 0.f);
      }
    }
    __syncthreads();
  }

  // ---- Stage G: f2 = drop(f @ W_ff2^T + b_ff2) + y -> global out (NT) ----
  v8f acc[8];
#pragma unroll
  for (int ti = 0; ti < 8; ++ti) acc[ti] = (v8f){};
  for (int kh = 0; kh < 2; ++kh) {
    __syncthreads();
    for (int i = tid; i < DM * 64; i += 256) {     // v4f units: 128 rows x 64 quads
      int n = i >> 6, q = i & 63;
      v4f w = *(const v4f*)(W_ff2 + (size_t)n * DFF + kh * 256 + q * 4);
      v4h p;
#pragma unroll
      for (int j = 0; j < 4; ++j) p[j] = (_Float16)w[j];
      *(v4h*)(sWst + (size_t)n * 256 + q * 4) = p;
    }
    __syncthreads();
    int n0 = wave << 4;
#pragma unroll
    for (int ti = 0; ti < 8; ++ti) {
      int m0 = ti << 4;
#pragma unroll
      for (int kc = 0; kc < 256; kc += 32)
        acc[ti] = wmma32(ldA16(sF, DFF, m0, kh * 256 + kc, lane),
                         ldA16(sWst, 256, n0, kc, lane), acc[ti]);
    }
  }
  {
    int n = (wave << 4) + (lane & 15);
    float bias = b_ff2[n];
    const float* m2b = m2 + (size_t)b * SEQL * DM;
    float* ob = out + (size_t)b * SEQL * DM;
#pragma unroll
    for (int ti = 0; ti < 8; ++ti) {
      int rb = (ti << 4) + ((lane >> 4) << 3);
#pragma unroll
      for (int v = 0; v < 8; ++v) {
        int r = rb + v;
        float f2 = acc[ti][v] + bias;
        f2 = (__builtin_nontemporal_load(m2b + r * DM + n) > DROP_P) ? f2 * DROP_SC : 0.f;
        __builtin_nontemporal_store(f2 + sH0[r * DM + n], ob + r * DM + n);
      }
    }
  }
}

extern "C" void kernel_launch(void* const* d_in, const int* in_sizes, int n_in,
                              void* d_out, int out_size, void* d_ws, size_t ws_size,
                              hipStream_t stream) {
  (void)in_sizes; (void)n_in; (void)out_size; (void)d_ws; (void)ws_size;
  const float* x         = (const float*)d_in[0];
  const float* W_embed   = (const float*)d_in[1];
  const float* b_embed   = (const float*)d_in[2];
  const float* g0        = (const float*)d_in[3];
  const float* b0        = (const float*)d_in[4];
  const float* W_qkv     = (const float*)d_in[5];
  const float* b_qkv     = (const float*)d_in[6];
  const float* attn_bias = (const float*)d_in[7];
  const float* key_bias  = (const float*)d_in[8];
  const float* W_out     = (const float*)d_in[9];
  const float* b_out     = (const float*)d_in[10];
  const float* g1        = (const float*)d_in[11];
  const float* b1        = (const float*)d_in[12];
  const float* g2        = (const float*)d_in[13];
  const float* b2        = (const float*)d_in[14];
  const float* W_ff1     = (const float*)d_in[15];
  const float* b_ff1     = (const float*)d_in[16];
  const float* W_ff2     = (const float*)d_in[17];
  const float* b_ff2     = (const float*)d_in[18];
  const float* m_attn    = (const float*)d_in[19];
  const float* m1        = (const float*)d_in[20];
  const float* m_ff      = (const float*)d_in[21];
  const float* m2        = (const float*)d_in[22];

  const size_t smem_bytes = 98304 + 196608;   // 288 KB dynamic (of 320 KB WGP LDS)
  hipLaunchKernelGGL(fused_block, dim3(BATCH), dim3(256), smem_bytes, stream,
                     x, W_embed, b_embed, g0, b0, W_qkv, b_qkv, attn_bias, key_bias,
                     W_out, b_out, g1, b1, g2, b2, W_ff1, b_ff1, W_ff2, b_ff2,
                     m_attn, m1, m_ff, m2, (float*)d_out);
}
